// NAM_classification_12343736009401
// MI455X (gfx1250) — matrix operand order repaired and checked by reading, not compiled, not run on hardware
//
#include <hip/hip_runtime.h>
#include <hip/hip_bf16.h>

typedef __attribute__((ext_vector_type(16))) _Float16 v16h;
typedef __attribute__((ext_vector_type(8)))  float    v8f;

#define NB     8192   // batch
#define NF     256    // features
#define NH     100    // hidden (padded to 128)
#define NO     50     // fc2 out (padded to 64)
#define ROWS   64     // batch rows per block (4 waves x 16)
#define FPB    16     // features per block
#define W2PAD  (64 * 128)   // halfs per packed W2 feature tile (16 KB)

#define NAM_ASYNC 1   // CDNA5 async global->LDS copy path

// ---------------------------------------------------------------------------
// async copy of one 16 KB W2 f16 tile into LDS: 128 threads x 8 x B128
// ---------------------------------------------------------------------------
__device__ __forceinline__ void copy_w2_tile(_Float16* dst_lds,
                                             const _Float16* src_g, int tid) {
#pragma unroll
    for (int r = 0; r < 8; ++r) {
        const int off = r * 2048 + tid * 16;   // bytes
#if NAM_ASYNC
        unsigned int lds_addr = (unsigned int)(unsigned long long)
                                ((char*)dst_lds + off);
        const char* g = (const char*)src_g + off;
        asm volatile("global_load_async_to_lds_b128 %0, %1, off"
                     :: "v"(lds_addr), "v"(g) : "memory");
#else
        *(uint4*)((char*)dst_lds + off) = *(const uint4*)((const char*)src_g + off);
#endif
    }
}

__device__ __forceinline__ void wait_async_zero() {
#if NAM_ASYNC
    asm volatile("s_wait_asynccnt 0x0" ::: "memory");
#endif
}

// A-layout K index for (hlf, kc, j): the permutation the WMMA A registers use
__host__ __device__ __forceinline__ int a_k_index(int hlf, int kc, int j) {
    return kc * 32 + hlf * 16 + j;
}

// ---------------------------------------------------------------------------
// zero the per-row accumulator (d_ws is poisoned by the harness)
// ---------------------------------------------------------------------------
__global__ void nam_zero_kernel(float* __restrict__ acc, int n) {
    int i = blockIdx.x * blockDim.x + threadIdx.x;
    if (i < n) acc[i] = 0.0f;
}

// ---------------------------------------------------------------------------
// pack W2 -> f16, zero-padded, in WMMA B layout: [f][o(64)][k(128)]
// ---------------------------------------------------------------------------
__global__ void nam_pack_w2_kernel(const float* __restrict__ W2,
                                   _Float16* __restrict__ w2p) {
    const int f = blockIdx.x;
    for (int i = threadIdx.x; i < W2PAD; i += blockDim.x) {
        int o = i >> 7, k = i & 127;
        float v = (o < NO && k < NH) ? W2[(size_t)f * (NO * NH) + o * NH + k]
                                     : 0.0f;
        w2p[(size_t)f * W2PAD + i] = (_Float16)v;
    }
}

// ---------------------------------------------------------------------------
// pack W1/b1 -> f16 in A-consumption order [f][hlf(2)][kc(4)][j(16)],
// and b2/W3 padded f32 (50 -> 64)
// ---------------------------------------------------------------------------
__global__ void nam_pack_small_kernel(const float* __restrict__ W1,
                                      const float* __restrict__ b1,
                                      const float* __restrict__ b2,
                                      const float* __restrict__ W3,
                                      _Float16* __restrict__ w1h,
                                      _Float16* __restrict__ b1h,
                                      float* __restrict__ b2p,
                                      float* __restrict__ w3p) {
    const int f = blockIdx.x;
    const int t = threadIdx.x;   // 128
    const int hlf = t >> 6, rem = t & 63, kc = rem >> 4, j = rem & 15;
    const int k = a_k_index(hlf, kc, j);
    w1h[f * 128 + t] = (k < NH) ? (_Float16)W1[(size_t)f * NH + k] : (_Float16)0.0f;
    b1h[f * 128 + t] = (k < NH) ? (_Float16)b1[(size_t)f * NH + k] : (_Float16)0.0f;
    if (t < 64) {
        b2p[f * 64 + t] = (t < NO) ? b2[(size_t)f * NO + t] : 0.0f;
        w3p[f * 64 + t] = (t < NO) ? W3[(size_t)f * NO + t] : 0.0f;
    }
}

// ---------------------------------------------------------------------------
// fused NAM main kernel: per-feature MLP via WMMA, async double-buffered W2
// grid: (NB/ROWS, NF/FPB), block: 128 threads (4 waves)
// ---------------------------------------------------------------------------
__global__ __launch_bounds__(128)
void nam_main_kernel(const float* __restrict__ x,      // [NB,NF]
                     const _Float16* __restrict__ w2p, // packed B tiles
                     const _Float16* __restrict__ w1h, // [NF,128] A-order f16
                     const _Float16* __restrict__ b1h, // [NF,128] A-order f16
                     const float* __restrict__ b2p,    // [NF,64]
                     const float* __restrict__ w3p,    // [NF,64]
                     float* __restrict__ acc_out)      // [NB]
{
    __shared__ alignas(32) _Float16 w2buf[2][W2PAD];   // 2 x 16 KB double buffer
    __shared__ float xs[ROWS * FPB];                   // 4 KB
    __shared__ alignas(32) _Float16 w1s[FPB * 128];    // 4 KB
    __shared__ alignas(32) _Float16 b1s[FPB * 128];    // 4 KB
    __shared__ float b2s[FPB * 64];                    // 4 KB
    __shared__ float w3s[FPB * 64];                    // 4 KB

    const int tid  = threadIdx.x;
    const int lane = tid & 31;
    const int wave = tid >> 5;           // 0..3
    const int m16  = lane & 15;          // row within 16-row tile / N column
    const int hlf  = lane >> 4;          // 0 or 1 (lane half)
    const int bbase = hlf ? 16 : 0;      // B-layout K base

    const int row0 = blockIdx.x * ROWS;
    const int f0   = blockIdx.y * FPB;

    // ---- block-start staging (vectorized) ----
    {   // x tile: 1024 floats = 256 uint4
        const uint4* src = (const uint4*)(x + (size_t)row0 * NF + f0);
        // row r occupies 4 uint4 within a 256-float row of x
        for (int i = tid; i < ROWS * 4; i += 128) {
            int r = i >> 2, c = i & 3;
            ((uint4*)xs)[i] = src[(size_t)r * (NF / 4) + c];
        }
    }
    {   // w1/b1: 2048 halfs = 256 uint4 each
        const uint4* s1 = (const uint4*)(w1h + (size_t)f0 * 128);
        const uint4* s2 = (const uint4*)(b1h + (size_t)f0 * 128);
        for (int i = tid; i < FPB * 128 / 8; i += 128) {
            ((uint4*)w1s)[i] = s1[i];
            ((uint4*)b1s)[i] = s2[i];
        }
    }
    {   // b2/w3: 1024 floats = 256 uint4 each
        const uint4* s1 = (const uint4*)(b2p + (size_t)f0 * 64);
        const uint4* s2 = (const uint4*)(w3p + (size_t)f0 * 64);
        for (int i = tid; i < FPB * 64 / 4; i += 128) {
            ((uint4*)b2s)[i] = s1[i];
            ((uint4*)w3s)[i] = s2[i];
        }
    }
    // prefetch feature 0's B tile (async into buffer 0)
    copy_w2_tile(w2buf[0], w2p + (size_t)f0 * W2PAD, tid);

    float racc[8];
#pragma unroll
    for (int r = 0; r < 8; ++r) racc[r] = 0.0f;

    for (int fi = 0; fi < FPB; ++fi) {
        const int cur = fi & 1;

        wait_async_zero();   // my wave's async loads into w2buf[cur] done
        __syncthreads();     // all waves done; previous buffer's readers done

        if (fi + 1 < FPB)    // overlap: stream next feature's B tile
            copy_w2_tile(w2buf[cur ^ 1],
                         w2p + (size_t)(f0 + fi + 1) * W2PAD, tid);

        const float xv = xs[(wave * 16 + m16) * FPB + fi];
        const _Float16 xh = (_Float16)xv;
        v16h xvv;
#pragma unroll
        for (int j = 0; j < 16; ++j) xvv[j] = xh;

        const _Float16* w1f = &w1s[fi * 128 + hlf * 64];
        const _Float16* b1f = &b1s[fi * 128 + hlf * 64];
        const _Float16* w2f = w2buf[cur];

        v8f acc[4];
#pragma unroll
        for (int nt = 0; nt < 4; ++nt) acc[nt] = (v8f){};

        // h2[16x64] = h1[16x128] * W2^T[128x64], K in 4 chunks of 32
#pragma unroll
        for (int kc = 0; kc < 4; ++kc) {
            // A (h1) tile in f16, built in packed-f16 math, A-layout order
            const v16h wv = *(const v16h*)&w1f[kc * 16];
            const v16h bv = *(const v16h*)&b1f[kc * 16];
            v16h hv = xvv * wv + bv;                 // v_pk_fma_f16
            v16h a;
#pragma unroll
            for (int j = 0; j < 16; ++j) {           // relu -> v_pk_max_num_f16
                _Float16 h = hv[j];
                a[j] = h > (_Float16)0.0f ? h : (_Float16)0.0f;
            }
#pragma unroll
            for (int nt = 0; nt < 4; ++nt) {
                const v16h bm = *(const v16h*)&w2f[(nt * 16 + m16) * 128 +
                                                  kc * 32 + bbase];
                acc[nt] = __builtin_amdgcn_wmma_f32_16x16x32_f16(
                    false, a, false, bm, (short)0, acc[nt], false, false);
            }
        }

        // epilogue: relu(h2 + b2) . W3, accumulated per batch row (f32)
#pragma unroll
        for (int nt = 0; nt < 4; ++nt) {
            const int o = nt * 16 + m16;
            const float bb = b2s[fi * 64 + o];
            const float w3 = w3s[fi * 64 + o];       // 0 for padded o
#pragma unroll
            for (int r = 0; r < 8; ++r) {
                float h2 = acc[nt][r] + bb;
                h2 = h2 > 0.0f ? h2 : 0.0f;
                racc[r] = fmaf(h2, w3, racc[r]);
            }
        }
    }

    // reduce racc over the 16 N-lanes of each half-wave
#pragma unroll
    for (int r = 0; r < 8; ++r) {
        float v = racc[r];
        v += __shfl_xor(v, 8, 16);
        v += __shfl_xor(v, 4, 16);
        v += __shfl_xor(v, 2, 16);
        v += __shfl_xor(v, 1, 16);
        racc[r] = v;
    }
    // C layout: VGPR r holds M = r + 8*hlf; lane m16==0 owns the row sums
    if (m16 == 0) {
        const int rowb = row0 + wave * 16 + hlf * 8;
#pragma unroll
        for (int r = 0; r < 8; ++r) {
            unsafeAtomicAdd(&acc_out[rowb + r], racc[r]);  // global_atomic_add_f32
        }
    }
}

// ---------------------------------------------------------------------------
// final: sigmoid(sum + bias)
// ---------------------------------------------------------------------------
__global__ void nam_sigmoid_kernel(const float* __restrict__ acc,
                                   const float* __restrict__ bias,
                                   float* __restrict__ out, int n) {
    int i = blockIdx.x * blockDim.x + threadIdx.x;
    if (i < n) {
        float v = acc[i] + bias[0];
        out[i] = 1.0f / (1.0f + expf(-v));
    }
}

extern "C" void kernel_launch(void* const* d_in, const int* in_sizes, int n_in,
                              void* d_out, int out_size, void* d_ws, size_t ws_size,
                              hipStream_t stream) {
    const float* x    = (const float*)d_in[0];
    const float* W1   = (const float*)d_in[1];
    const float* b1   = (const float*)d_in[2];
    const float* W2   = (const float*)d_in[3];
    const float* b2   = (const float*)d_in[4];
    const float* W3   = (const float*)d_in[5];
    const float* bias = (const float*)d_in[6];
    float* out = (float*)d_out;

    // workspace layout
    char* ws = (char*)d_ws;
    float*    acc = (float*)ws;        ws += NB * sizeof(float);          // 32 KB
    _Float16* w2p = (_Float16*)ws;     ws += (size_t)NF * W2PAD * 2;      // 4 MB
    _Float16* w1h = (_Float16*)ws;     ws += NF * 128 * 2;                // 64 KB
    _Float16* b1h = (_Float16*)ws;     ws += NF * 128 * 2;                // 64 KB
    float*    b2p = (float*)ws;        ws += NF * 64 * sizeof(float);     // 64 KB
    float*    w3p = (float*)ws;        ws += NF * 64 * sizeof(float);     // 64 KB

    nam_zero_kernel<<<(NB + 255) / 256, 256, 0, stream>>>(acc, NB);
    nam_pack_w2_kernel<<<NF, 256, 0, stream>>>(W2, w2p);
    nam_pack_small_kernel<<<NF, 128, 0, stream>>>(W1, b1, b2, W3,
                                                  w1h, b1h, b2p, w3p);

    dim3 grid(NB / ROWS, NF / FPB);   // 128 x 16
    nam_main_kernel<<<grid, 128, 0, stream>>>(x, w2p, w1h, b1h, b2p, w3p, acc);

    nam_sigmoid_kernel<<<(NB + 255) / 256, 256, 0, stream>>>(acc, bias, out, NB);
}
